// GN_13125420057113
// MI455X (gfx1250) — matrix operand order repaired and checked by reading, not compiled, not run on hardware
//
#include <hip/hip_runtime.h>

#define NNODES  50000
#define NEDGES  600000
#define NGRAPHS 64
#define DD      128
#define NCLS    10
#define NPASS   3
#define NT      8           // 8 column tiles of 16 -> 128 outputs

typedef __attribute__((ext_vector_type(16))) __bf16       v16bf;
typedef __attribute__((ext_vector_type(8)))  float        v8f;
typedef __attribute__((ext_vector_type(4)))  unsigned int u32x4;

union Frag16 { v16bf v; u32x4 q[2]; };

__device__ __forceinline__ unsigned short f32_to_bf16(float f) {
  unsigned u = __float_as_uint(f);
  u += 0x7FFFu + ((u >> 16) & 1u);           // round-to-nearest-even
  return (unsigned short)(u >> 16);
}

// ---------------- small utility kernels ----------------

__global__ void zero_f32_kernel(float* p, int n) {
  for (int i = blockIdx.x * blockDim.x + threadIdx.x; i < n; i += gridDim.x * blockDim.x)
    p[i] = 0.f;
}

__global__ void cvt_f32_bf16_kernel(const float* __restrict__ s,
                                    unsigned short* __restrict__ d, int n) {
  for (int i = blockIdx.x * blockDim.x + threadIdx.x; i < n; i += gridDim.x * blockDim.x)
    d[i] = f32_to_bf16(s[i]);
}

__global__ void copy_f32_kernel(const float* __restrict__ s, float* __restrict__ d, int n) {
  for (int i = blockIdx.x * blockDim.x + threadIdx.x; i < n; i += gridDim.x * blockDim.x)
    d[i] = s[i];
}

__global__ void count_edges_kernel(const int* __restrict__ dstI, float* __restrict__ cnt) {
  for (int e = blockIdx.x * blockDim.x + threadIdx.x; e < NEDGES; e += gridDim.x * blockDim.x)
    atomicAdd(&cnt[dstI[e]], 1.f);
}

__global__ void count_nodes_kernel(const int* __restrict__ batchI, float* __restrict__ cnt) {
  for (int n = blockIdx.x * blockDim.x + threadIdx.x; n < NNODES; n += gridDim.x * blockDim.x)
    atomicAdd(&cnt[batchI[n]], 1.f);
}

// edge_bar = acc / max(cnt,1); write bf16 copy; accumulate per-graph sums
__global__ void normalize_ebar_kernel(const float* __restrict__ acc,
                                      const float* __restrict__ cnt,
                                      const int* __restrict__ batchI,
                                      unsigned short* __restrict__ ebarb,
                                      float* __restrict__ eacc_g) {
  const int total = NNODES * DD;
  for (int i = blockIdx.x * blockDim.x + threadIdx.x; i < total; i += gridDim.x * blockDim.x) {
    int node = i >> 7, c = i & (DD - 1);
    float v = acc[i] / fmaxf(cnt[node], 1.f);
    ebarb[i] = f32_to_bf16(v);
    atomicAdd(&eacc_g[(size_t)batchI[node] * DD + c], v);
  }
}

// u' = relu([node_mean, eb_mean, u] @ Wg + bg)   (64x128 outputs, trivial)
__global__ void global_update_kernel(const float* __restrict__ nacc,
                                     const float* __restrict__ eacc,
                                     const float* __restrict__ cntg,
                                     const float* __restrict__ ucur,
                                     const float* __restrict__ Wg_w,
                                     const float* __restrict__ Wg_b,
                                     float* __restrict__ unext,
                                     unsigned short* __restrict__ ubf) {
  int idx = blockIdx.x * blockDim.x + threadIdx.x;
  if (idx >= NGRAPHS * DD) return;
  int g = idx >> 7, c = idx & (DD - 1);
  float rc = 1.f / fmaxf(cntg[g], 1.f);
  const float* nrow = nacc + (size_t)g * DD;
  const float* erow = eacc + (size_t)g * DD;
  const float* urow = ucur + (size_t)g * DD;
  float acc = Wg_b[c];
  for (int k = 0; k < DD; ++k) acc += (nrow[k] * rc) * Wg_w[k * DD + c];
  for (int k = 0; k < DD; ++k) acc += (erow[k] * rc) * Wg_w[(DD + k) * DD + c];
  for (int k = 0; k < DD; ++k) acc += urow[k] * Wg_w[(2 * DD + k) * DD + c];
  float v = fmaxf(acc, 0.f);
  unext[idx] = v;
  ubf[idx] = f32_to_bf16(v);
}

// out = pooled @ lin_w + lin_b   (64x10 outputs)
__global__ void readout_kernel(const float* __restrict__ nacc,
                               const float* __restrict__ eacc,
                               const float* __restrict__ cntg,
                               const float* __restrict__ ufin,
                               const float* __restrict__ lw,
                               const float* __restrict__ lb,
                               float* __restrict__ out) {
  int idx = blockIdx.x * blockDim.x + threadIdx.x;
  if (idx >= NGRAPHS * NCLS) return;
  int g = idx / NCLS, c = idx % NCLS;
  float rc = 1.f / fmaxf(cntg[g], 1.f);
  float acc = lb[c];
  for (int k = 0; k < DD; ++k) acc += (nacc[(size_t)g * DD + k] * rc) * lw[k * NCLS + c];
  for (int k = 0; k < DD; ++k) acc += (eacc[(size_t)g * DD + k] * rc) * lw[(DD + k) * NCLS + c];
  for (int k = 0; k < DD; ++k) acc += ufin[(size_t)g * DD + k] * lw[(2 * DD + k) * NCLS + c];
  out[idx] = acc;
}

// ---------------- WMMA edge update ----------------
// e' = relu([n_src | n_dst | e | u[b(src)]] @ We + be), plus scatter-add into
// edge_bar accumulator at dst. One wave = one 16-edge x 128-col tile.
// KT=16 k-steps of 32 (4 segments x 128). B fragments pre-swizzled into LDS.
// Plain launch_bounds(256): letting the allocator use >256 VGPRs (MSB
// addressing) avoids scratch spills and keeps the compiler's deep VMEM
// pipelining; LDS already caps blocks/WGP at 2.
__global__ __launch_bounds__(256) void edge_update_wmma(
    const unsigned short* __restrict__ nodeb,
    unsigned short* __restrict__ edgeb,
    const unsigned short* __restrict__ ub,
    const float* __restrict__ Ww,     // [4D, D]
    const float* __restrict__ Wb,     // [D]
    const int* __restrict__ srcI,
    const int* __restrict__ dstI,
    const int* __restrict__ batchI,
    float* __restrict__ ebar_acc,
    int nTiles) {
  extern __shared__ unsigned short Bsh[];
  const int KT = 16;
  const int tid = threadIdx.x;

  // Pre-swizzle B: entry (kStep, nTile, lane) holds the lane's 16 bf16 values
  // per ISA B layout: lane = column (lane&15); lanes>=16 carry K=8..15/24..31.
  const int totalEnt = KT * NT * 32;
  for (int idx = tid; idx < totalEnt; idx += blockDim.x) {
    int f = idx >> 5, lane = idx & 31;
    int kStep = f >> 3, nt = f & 7;
    int n = nt * 16 + (lane & 15);
    int kbase = kStep * 32 + ((lane >= 16) ? 8 : 0);
    unsigned short* d = &Bsh[idx * 16];
#pragma unroll
    for (int j = 0; j < 16; ++j) d[j] = f32_to_bf16(Ww[(kbase + j + (j >= 8 ? 8 : 0)) * DD + n]);
  }
  __syncthreads();

  const int lane = tid & 31;
  const int wid = tid >> 5;
  const int halfSel = lane >> 4;     // A: lanes>=16 hold K-halves 8..15/24..31
  const int rowInTile = lane & 15;
  const int aK0 = halfSel ? 8 : 0;
  const int wavesPerGrid = gridDim.x * (blockDim.x >> 5);

  for (int t = blockIdx.x * (blockDim.x >> 5) + wid; t < nTiles; t += wavesPerGrid) {
    const int e0 = t * 16;
    const int myE = e0 + rowInTile;
    const int mySrc = srcI[myE];
    const int myDst = dstI[myE];
    const int myB = batchI[mySrc];
    const unsigned short* segPtr[4] = {
        nodeb + (size_t)mySrc * DD,
        nodeb + (size_t)myDst * DD,
        edgeb + (size_t)myE * DD,
        ub + (size_t)myB * DD};

    v8f acc[NT] = {};
#pragma unroll
    for (int seg = 0; seg < 4; ++seg) {
      const unsigned short* p = segPtr[seg];
#pragma unroll
      for (int kk = 0; kk < 4; ++kk) {
        const int kloc = kk * 32 + aK0;
        Frag16 a;
        a.q[0] = *(const u32x4*)(p + kloc);        // K = kloc..kloc+7
        a.q[1] = *(const u32x4*)(p + kloc + 16);   // K = kloc+16..kloc+23
        const int kStep = seg * 4 + kk;
#pragma unroll
        for (int nt = 0; nt < NT; ++nt) {
          Frag16 b;
          const unsigned short* bp = Bsh + ((size_t)(kStep * NT + nt) * 32 + lane) * 16;
          b.q[0] = ((const u32x4*)bp)[0];
          b.q[1] = ((const u32x4*)bp)[1];
          acc[nt] = __builtin_amdgcn_wmma_f32_16x16x32_bf16(
              false, a.v, false, b.v, (short)0, acc[nt], false, false);
        }
      }
    }

    // C layout: lane holds column (lane&15); VGPR r -> row r + 8*halfSel
    int rowDst[8];
#pragma unroll
    for (int r = 0; r < 8; ++r) rowDst[r] = dstI[e0 + halfSel * 8 + r];
#pragma unroll
    for (int nt = 0; nt < NT; ++nt) {
      const int col = nt * 16 + rowInTile;
      const float bias = Wb[col];
#pragma unroll
      for (int r = 0; r < 8; ++r) {
        const int m = halfSel * 8 + r;
        float v = acc[nt][r] + bias;
        v = v > 0.f ? v : 0.f;
        edgeb[(size_t)(e0 + m) * DD + col] = f32_to_bf16(v);
        atomicAdd(&ebar_acc[(size_t)rowDst[r] * DD + col], v);
      }
    }
  }
}

// ---------------- WMMA node update ----------------
// n' = relu([n | edge_bar | u[b(n)]] @ Wn + bn), plus per-graph sum scatter.
__global__ __launch_bounds__(256) void node_update_wmma(
    unsigned short* __restrict__ nodeb,
    const unsigned short* __restrict__ ebarb,
    const unsigned short* __restrict__ ub,
    const float* __restrict__ Ww,     // [3D, D]
    const float* __restrict__ Wb,     // [D]
    const int* __restrict__ batchI,
    float* __restrict__ ngraph_acc,
    int nTiles) {
  extern __shared__ unsigned short Bsh[];
  const int KT = 12;
  const int tid = threadIdx.x;

  const int totalEnt = KT * NT * 32;
  for (int idx = tid; idx < totalEnt; idx += blockDim.x) {
    int f = idx >> 5, lane = idx & 31;
    int kStep = f >> 3, nt = f & 7;
    int n = nt * 16 + (lane & 15);
    int kbase = kStep * 32 + ((lane >= 16) ? 8 : 0);
    unsigned short* d = &Bsh[idx * 16];
#pragma unroll
    for (int j = 0; j < 16; ++j) d[j] = f32_to_bf16(Ww[(kbase + j + (j >= 8 ? 8 : 0)) * DD + n]);
  }
  __syncthreads();

  const int lane = tid & 31;
  const int wid = tid >> 5;
  const int halfSel = lane >> 4;
  const int rowInTile = lane & 15;
  const int aK0 = halfSel ? 8 : 0;
  const int wavesPerGrid = gridDim.x * (blockDim.x >> 5);

  for (int t = blockIdx.x * (blockDim.x >> 5) + wid; t < nTiles; t += wavesPerGrid) {
    const int n0 = t * 16;
    const int myN = n0 + rowInTile;
    const int myB = batchI[myN];
    const unsigned short* segPtr[3] = {
        nodeb + (size_t)myN * DD,
        ebarb + (size_t)myN * DD,
        ub + (size_t)myB * DD};

    v8f acc[NT] = {};
#pragma unroll
    for (int seg = 0; seg < 3; ++seg) {
      const unsigned short* p = segPtr[seg];
#pragma unroll
      for (int kk = 0; kk < 4; ++kk) {
        const int kloc = kk * 32 + aK0;
        Frag16 a;
        a.q[0] = *(const u32x4*)(p + kloc);
        a.q[1] = *(const u32x4*)(p + kloc + 16);
        const int kStep = seg * 4 + kk;
#pragma unroll
        for (int nt = 0; nt < NT; ++nt) {
          Frag16 b;
          const unsigned short* bp = Bsh + ((size_t)(kStep * NT + nt) * 32 + lane) * 16;
          b.q[0] = ((const u32x4*)bp)[0];
          b.q[1] = ((const u32x4*)bp)[1];
          acc[nt] = __builtin_amdgcn_wmma_f32_16x16x32_bf16(
              false, a.v, false, b.v, (short)0, acc[nt], false, false);
        }
      }
    }

    int rowG[8];
#pragma unroll
    for (int r = 0; r < 8; ++r) rowG[r] = batchI[n0 + halfSel * 8 + r];
#pragma unroll
    for (int nt = 0; nt < NT; ++nt) {
      const int col = nt * 16 + rowInTile;
      const float bias = Wb[col];
#pragma unroll
      for (int r = 0; r < 8; ++r) {
        const int m = halfSel * 8 + r;
        float v = acc[nt][r] + bias;
        v = v > 0.f ? v : 0.f;
        nodeb[(size_t)(n0 + m) * DD + col] = f32_to_bf16(v);
        atomicAdd(&ngraph_acc[(size_t)rowG[r] * DD + col], v);
      }
    }
  }
}

// ---------------- host orchestration ----------------

static inline size_t align_up(size_t x) { return (x + 255) & ~(size_t)255; }

extern "C" void kernel_launch(void* const* d_in, const int* in_sizes, int n_in,
                              void* d_out, int out_size, void* d_ws, size_t ws_size,
                              hipStream_t stream) {
  const float* node_f = (const float*)d_in[0];
  const float* edge_f = (const float*)d_in[1];
  const float* u_f    = (const float*)d_in[2];
  const int*   eidx   = (const int*)d_in[3];
  const int*   srcI   = eidx;
  const int*   dstI   = eidx + NEDGES;
  const int*   batchI = (const int*)d_in[4];
  const float* We_w = (const float*)d_in[5];
  const float* We_b = (const float*)d_in[6];
  const float* Wn_w = (const float*)d_in[7];
  const float* Wn_b = (const float*)d_in[8];
  const float* Wg_w = (const float*)d_in[9];
  const float* Wg_b = (const float*)d_in[10];
  const float* lin_w = (const float*)d_in[11];
  const float* lin_b = (const float*)d_in[12];
  float* out = (float*)d_out;

  // workspace carve-up
  char* base = (char*)d_ws;
  size_t off = 0;
  auto take = [&](size_t bytes) -> char* {
    char* p = base + off;
    off = align_up(off + bytes);
    return p;
  };
  unsigned short* edgeb   = (unsigned short*)take((size_t)NEDGES * DD * 2);
  unsigned short* nodeb   = (unsigned short*)take((size_t)NNODES * DD * 2);
  float*          ebarAcc = (float*)take((size_t)NNODES * DD * 4);
  unsigned short* ebarb   = (unsigned short*)take((size_t)NNODES * DD * 2);
  float*          cntN    = (float*)take((size_t)NNODES * 4);
  float*          cntG    = (float*)take((size_t)NGRAPHS * 4);
  float*          nAccG   = (float*)take((size_t)NGRAPHS * DD * 4);
  float*          eAccG   = (float*)take((size_t)NGRAPHS * DD * 4);
  float*          uA      = (float*)take((size_t)NGRAPHS * DD * 4);
  float*          uB      = (float*)take((size_t)NGRAPHS * DD * 4);
  unsigned short* ubf     = (unsigned short*)take((size_t)NGRAPHS * DD * 2);

  const int T = 256;
  auto blocks = [&](int n) { int b = (n + T - 1) / T; return b > 16384 ? 16384 : b; };

  // ---- one-time prep ----
  cvt_f32_bf16_kernel<<<blocks(NNODES * DD), T, 0, stream>>>(node_f, nodeb, NNODES * DD);
  cvt_f32_bf16_kernel<<<blocks(NEDGES * DD), T, 0, stream>>>(edge_f, edgeb, NEDGES * DD);
  cvt_f32_bf16_kernel<<<blocks(NGRAPHS * DD), T, 0, stream>>>(u_f, ubf, NGRAPHS * DD);
  copy_f32_kernel<<<blocks(NGRAPHS * DD), T, 0, stream>>>(u_f, uA, NGRAPHS * DD);
  zero_f32_kernel<<<blocks(NNODES), T, 0, stream>>>(cntN, NNODES);
  zero_f32_kernel<<<1, T, 0, stream>>>(cntG, NGRAPHS);
  count_edges_kernel<<<blocks(NEDGES), T, 0, stream>>>(dstI, cntN);
  count_nodes_kernel<<<blocks(NNODES), T, 0, stream>>>(batchI, cntG);

  const int edgeTiles = NEDGES / 16;   // 37500, exact
  const int nodeTiles = NNODES / 16;   // 3125, exact
  const size_t edgeShmem = 16 * NT * 32 * 16 * sizeof(unsigned short);  // 128 KB
  const size_t nodeShmem = 12 * NT * 32 * 16 * sizeof(unsigned short);  //  96 KB

  float* uCur = uA;
  float* uNxt = uB;
  for (int p = 0; p < NPASS; ++p) {
    zero_f32_kernel<<<blocks(NNODES * DD), T, 0, stream>>>(ebarAcc, NNODES * DD);
    zero_f32_kernel<<<blocks(NGRAPHS * DD), T, 0, stream>>>(nAccG, NGRAPHS * DD);
    zero_f32_kernel<<<blocks(NGRAPHS * DD), T, 0, stream>>>(eAccG, NGRAPHS * DD);

    edge_update_wmma<<<600, T, edgeShmem, stream>>>(
        nodeb, edgeb, ubf, We_w, We_b, srcI, dstI, batchI, ebarAcc, edgeTiles);

    normalize_ebar_kernel<<<blocks(NNODES * DD), T, 0, stream>>>(
        ebarAcc, cntN, batchI, ebarb, eAccG);

    node_update_wmma<<<256, T, nodeShmem, stream>>>(
        nodeb, ebarb, ubf, Wn_w, Wn_b, batchI, nAccG, nodeTiles);

    global_update_kernel<<<(NGRAPHS * DD + T - 1) / T, T, 0, stream>>>(
        nAccG, eAccG, cntG, uCur, Wg_w, Wg_b, uNxt, ubf);

    float* tmp = uCur; uCur = uNxt; uNxt = tmp;
  }

  // final u is in uCur after the swap
  readout_kernel<<<(NGRAPHS * NCLS + T - 1) / T, T, 0, stream>>>(
      nAccG, eAccG, cntG, uCur, lin_w, lin_b, out);
}
